// ConvAttnPool_68410239091516
// MI455X (gfx1250) — matrix operand order repaired and checked by reading, not compiled, not run on hardware
//
#include <hip/hip_runtime.h>
#include <hip/hip_bf16.h>
#include <math.h>

typedef __attribute__((ext_vector_type(16))) _Float16 v16h;
typedef __attribute__((ext_vector_type(8)))  _Float16 h8;
typedef __attribute__((ext_vector_type(8)))  float    v8f;

#define B_   8
#define L_   2500
#define E_   100
#define D_   128
#define Y_   8921
#define K_   9
#define PAD_ 4
#define TL   32      // sequence tile for the conv kernel
#define LP   2560    // padded sequence length (multiple of 64, pad rows are zero)
#define YP   8928    // padded label count (multiple of 32, pad rows are zero)
#define LPAD 40      // padded l-stride (halfs) of transposed LDS h-tile (80B)

// ---------------------------------------------------------------------------
// Kernel 1: embedding gather + Conv1d(same) + tanh -> h_f16 [B][LP][D]
// (rows L..LP-1 written as zeros so downstream loads need no bounds checks)
// block = 128 threads (one per output channel d), grid = (LP/TL, B)
// ---------------------------------------------------------------------------
__global__ __launch_bounds__(128) void k_embed_conv(
    const int* __restrict__ x, const float* __restrict__ emb,
    const float* __restrict__ w /*[D][E][K]*/, const float* __restrict__ bias,
    _Float16* __restrict__ h16)
{
    __shared__ float se[(TL + K_ - 1) * E_];   // 40 * 100 floats = 16 KB
    const int l0  = blockIdx.x * TL;
    const int b   = blockIdx.y;
    const int tid = threadIdx.x;

    for (int idx = tid; idx < (TL + K_ - 1) * E_; idx += 128) {
        const int r = idx / E_;
        const int e = idx - r * E_;
        const int lp = l0 - PAD_ + r;
        float v = 0.0f;
        if (lp >= 0 && lp < L_) {
            const int tok = x[b * L_ + lp];
            v = emb[(size_t)tok * E_ + e];     // row 0 is zero (padding_idx)
        }
        se[idx] = v;
    }
    __syncthreads();

    const int d = tid;
    float acc[TL];
    const float bb = bias[d];
#pragma unroll
    for (int l = 0; l < TL; ++l) acc[l] = bb;

    const float* wd = w + (size_t)d * (E_ * K_);
    for (int e = 0; e < E_; ++e) {
        float col[TL + K_ - 1];
#pragma unroll
        for (int i = 0; i < TL + K_ - 1; ++i) col[i] = se[i * E_ + e];
#pragma unroll
        for (int k = 0; k < K_; ++k) {
            const float wv = wd[e * K_ + k];
#pragma unroll
            for (int l = 0; l < TL; ++l) acc[l] = fmaf(col[l + k], wv, acc[l]);
        }
    }
#pragma unroll
    for (int l = 0; l < TL; ++l) {
        const int lp = l0 + l;   // lp < LP by construction
        h16[((size_t)b * LP + lp) * D_ + d] =
            (lp < L_) ? (_Float16)tanhf(acc[l]) : (_Float16)0.0f;
    }
}

// ---------------------------------------------------------------------------
// Kernel 2: U (f32 [Y][D]) -> U16 (f16 [YP][D]), pad rows zeroed
// ---------------------------------------------------------------------------
__global__ void k_u_to_f16(const float* __restrict__ src,
                           _Float16* __restrict__ dst)
{
    const int i = blockIdx.x * blockDim.x + threadIdx.x;
    if (i < YP * D_)
        dst[i] = (i < Y_ * D_) ? (_Float16)src[i] : (_Float16)0.0f;
}

// A-fragment loader: 16x32 f16 tile row, contiguous k order (k = 16*hi + s), same as B
__device__ __forceinline__ v16h load_afrag(const _Float16* rowp, int k0, int hi)
{
    const h8 c0 = *(const h8*)(rowp + k0 + hi * 16);
    const h8 c1 = *(const h8*)(rowp + k0 + hi * 16 + 8);
    v16h a;
#pragma unroll
    for (int j = 0; j < 8; ++j) { a[j] = c0[j]; a[j + 8] = c1[j]; }
    return a;
}

// ---------------------------------------------------------------------------
// Kernel 3: scores[b][y][l] = sum_d U16[y][d] * h16[b][l][d]   (WMMA)
// wave -> 32 y x 64 l (2x4 tiles): A reused 4x, B reused 2x, 32 WMMAs.
// All loads unconditional (padded operands); stores guarded with uniform
// fast path. block = 128 (4 waves), grid = (LP/64, ceil((YP/32)/4), B)
// ---------------------------------------------------------------------------
__global__ __launch_bounds__(128) void k_scores(
    const _Float16* __restrict__ U16, const _Float16* __restrict__ h16,
    float* __restrict__ scores /*[B][Y][L]*/)
{
    const int lane = threadIdx.x & 31;   // wave32
    const int wid  = threadIdx.x >> 5;
    const int row  = lane & 15;
    const int hi   = lane >> 4;          // 0 or 1

    const int l0 = blockIdx.x * 64;
    const int y0 = (blockIdx.y * 4 + wid) * 32;
    const int b  = blockIdx.z;
    if (y0 >= YP) return;                // uniform per-wave guard

    const _Float16* u0 = U16 + (size_t)(y0 + row) * D_;
    const _Float16* u1 = U16 + (size_t)(y0 + 16 + row) * D_;
    const _Float16* hbase = h16 + ((size_t)b * LP + l0 + row) * D_;

    v8f zero = {};
    v8f acc[2][4];
#pragma unroll
    for (int g = 0; g < 2; ++g)
#pragma unroll
        for (int s = 0; s < 4; ++s) acc[g][s] = zero;

#pragma unroll
    for (int k0 = 0; k0 < D_; k0 += 32) {
        const v16h a0 = load_afrag(u0, k0, hi);
        const v16h a1 = load_afrag(u1, k0, hi);
#pragma unroll
        for (int s = 0; s < 4; ++s) {
            // B fragment: col N=row, K=hi*16+j -> 32B contiguous, always in-bounds
            const v16h bf = *(const v16h*)(hbase + (size_t)(s * 16) * D_ + k0 + hi * 16);
            acc[0][s] = __builtin_amdgcn_wmma_f32_16x16x32_f16(
                            false, a0, false, bf, (short)0, acc[0][s], false, false);
            acc[1][s] = __builtin_amdgcn_wmma_f32_16x16x32_f16(
                            false, a1, false, bf, (short)0, acc[1][s], false, false);
        }
    }

    float* sb = scores + (size_t)b * Y_ * L_;
    if (y0 + 32 <= Y_ && l0 + 64 <= L_) {      // uniform fast path: no guards
#pragma unroll
        for (int g = 0; g < 2; ++g)
#pragma unroll
            for (int s = 0; s < 4; ++s) {
                const int ll = l0 + s * 16 + row;
#pragma unroll
                for (int r = 0; r < 8; ++r) {
                    const int yy = y0 + g * 16 + r + hi * 8;
                    __builtin_nontemporal_store(acc[g][s][r], sb + (size_t)yy * L_ + ll);
                }
            }
    } else {
#pragma unroll
        for (int g = 0; g < 2; ++g)
#pragma unroll
            for (int s = 0; s < 4; ++s) {
                const int ll = l0 + s * 16 + row;
#pragma unroll
                for (int r = 0; r < 8; ++r) {
                    const int yy = y0 + g * 16 + r + hi * 8;
                    if (yy < Y_ && ll < L_)
                        __builtin_nontemporal_store(acc[g][s][r], sb + (size_t)yy * L_ + ll);
                }
            }
    }
}

// ---------------------------------------------------------------------------
// Kernel 4: in-place softmax over L per (b,y) row; wave per row, row in LDS.
// block = 128 (4 waves, 40 KB LDS), grid = ceil(B*Y/4)
// ---------------------------------------------------------------------------
__global__ __launch_bounds__(128) void k_softmax(float* __restrict__ alpha)
{
    __shared__ float srow[4 * L_];       // 40000 bytes
    const int lane  = threadIdx.x & 31;
    const int wid   = threadIdx.x >> 5;
    const int rowid = blockIdx.x * 4 + wid;
    if (rowid >= B_ * Y_) return;

    float* p = alpha + (size_t)rowid * L_;
    float* s = srow + wid * L_;

    float mx = -3.0e38f;
    for (int i = lane; i < L_; i += 32) {
        const float v = __builtin_nontemporal_load(p + i);
        s[i] = v;
        mx = fmaxf(mx, v);
    }
#pragma unroll
    for (int o = 16; o > 0; o >>= 1) mx = fmaxf(mx, __shfl_xor(mx, o, 32));

    float sum = 0.0f;
    for (int i = lane; i < L_; i += 32) {
        const float e = __expf(s[i] - mx);
        s[i] = e;
        sum += e;
    }
#pragma unroll
    for (int o = 16; o > 0; o >>= 1) sum += __shfl_xor(sum, o, 32);

    const float inv = 1.0f / sum;
    for (int i = lane; i < L_; i += 32)
        __builtin_nontemporal_store(s[i] * inv, p + i);
}

// ---------------------------------------------------------------------------
// Kernel 5 helper: one 32-l chunk of the attention pooling GEMM.
// Stages h[32][128] transposed into sbuf ([d][l], padded stride), one
// barrier, A from the alpha stream (NT loads), 8 WMMAs from LDS.
// ---------------------------------------------------------------------------
template <bool TAIL>
__device__ __forceinline__ void mpool_step(
    const float* __restrict__ arow, const _Float16* __restrict__ hb,
    _Float16* __restrict__ sbuf, int l0, int tid, int row, int hi,
    v8f (&acc)[8])
{
    const int lrel = tid >> 2;           // 0..31
    const int dblk = (tid & 3) * 32;     // 0,32,64,96
    const _Float16* src = hb + (size_t)(l0 + lrel) * D_ + dblk;
    if (!TAIL) __builtin_prefetch(src + 32 * D_, 0, 1);     // next chunk of h
#pragma unroll
    for (int j = 0; j < 32; j += 8) {
        const h8 v = *(const h8*)(src + j);
#pragma unroll
        for (int q = 0; q < 8; ++q)
            sbuf[(dblk + j + q) * LPAD + lrel] = v[q];
    }
    __syncthreads();

    v16h a;
#pragma unroll
    for (int j = 0; j < 16; ++j) {
        const int k = hi * 16 + j;
        const int l = l0 + k;
        float v;
        if (TAIL) v = (l < L_) ? __builtin_nontemporal_load(arow + l) : 0.0f;
        else      v = __builtin_nontemporal_load(arow + l);
        a[j] = (_Float16)v;
    }
    if (!TAIL && l0 + 64 <= L_)
        __builtin_prefetch(arow + l0 + 32 + hi * 8, 0, 1);  // next alpha chunk

#pragma unroll
    for (int t = 0; t < 8; ++t) {
        const _Float16* bp = sbuf + (size_t)(t * 16 + row) * LPAD + hi * 16;
        const h8 b0 = *(const h8*)(bp);
        const h8 b1 = *(const h8*)(bp + 8);
        v16h bf;
#pragma unroll
        for (int j = 0; j < 8; ++j) { bf[j] = b0[j]; bf[j + 8] = b1[j]; }
        acc[t] = __builtin_amdgcn_wmma_f32_16x16x32_f16(
                     false, a, false, bf, (short)0, acc[t], false, false);
    }
}

// ---------------------------------------------------------------------------
// Kernel 5: m[b][y][d] = sum_l alpha[b][y][l] * h16[b][l][d]   (WMMA)
// block = 4 waves; wave -> 16 y x all 128 d. Double-buffered LDS tile,
// two barriers per full chunk, 78 unguarded chunks + 1 guarded tail.
// grid = (ceil(Yt/4), B)
// ---------------------------------------------------------------------------
__global__ __launch_bounds__(128) void k_mpool(
    const float* __restrict__ alpha, const _Float16* __restrict__ h16,
    float* __restrict__ m /*[B][Y][D]*/)
{
    __shared__ _Float16 sh[2][D_ * LPAD];   // 2 x 10240 B, [d][l] transposed
    const int tid  = threadIdx.x;
    const int lane = tid & 31;
    const int wid  = tid >> 5;
    const int row  = lane & 15;
    const int hi   = lane >> 4;

    const int ytiles = (Y_ + 15) / 16;
    const int yt = blockIdx.x * 4 + wid;
    const int b  = blockIdx.y;
    const bool wok = (yt < ytiles);
    const int y0 = yt * 16;
    const int yr = y0 + row;

    // clamp row: lanes past Y_ read row 0 (results discarded at store)
    const float* arow = alpha + ((size_t)b * Y_ + ((wok && yr < Y_) ? yr : 0)) * L_;
    const _Float16* hb = h16 + (size_t)b * LP * D_;

    v8f zero = {};
    v8f acc[8];
#pragma unroll
    for (int t = 0; t < 8; ++t) acc[t] = zero;

    const int nfull = L_ / 32;           // 78 full chunks
#pragma unroll 1
    for (int ci = 0; ci < nfull; ++ci) {
        mpool_step<false>(arow, hb, &sh[ci & 1][0], ci * 32, tid, row, hi, acc);
        __syncthreads();  // NOTE: redundant given next step's post-stage barrier
    }
    mpool_step<true>(arow, hb, &sh[nfull & 1][0], nfull * 32, tid, row, hi, acc);

    if (wok) {
        if (y0 + 16 <= Y_) {             // uniform fast path
#pragma unroll
            for (int t = 0; t < 8; ++t)
#pragma unroll
                for (int r = 0; r < 8; ++r)
                    m[((size_t)b * Y_ + y0 + r + hi * 8) * D_ + t * 16 + row] = acc[t][r];
        } else {
#pragma unroll
            for (int t = 0; t < 8; ++t)
#pragma unroll
                for (int r = 0; r < 8; ++r) {
                    const int yy = y0 + r + hi * 8;
                    if (yy < Y_)
                        m[((size_t)b * Y_ + yy) * D_ + t * 16 + row] = acc[t][r];
                }
        }
    }
}

// ---------------------------------------------------------------------------
// Kernel 6: logits = final_w . m + final_b ; yhat = sigmoid(logits)
// ---------------------------------------------------------------------------
__global__ __launch_bounds__(256) void k_logits(
    const float* __restrict__ m, const float* __restrict__ fw,
    const float* __restrict__ fb, float* __restrict__ logits,
    float* __restrict__ yhat)
{
    const int i = blockIdx.x * blockDim.x + threadIdx.x;
    if (i >= B_ * Y_) return;
    const int b = i / Y_;
    const int y = i - b * Y_;
    const float4* mp = (const float4*)(m + (size_t)i * D_);
    const float4* wp = (const float4*)(fw + (size_t)y * D_);
    float s = fb[y];
#pragma unroll 8
    for (int j = 0; j < D_ / 4; ++j) {
        const float4 a = mp[j];
        const float4 c = wp[j];
        s = fmaf(a.x, c.x, s);
        s = fmaf(a.y, c.y, s);
        s = fmaf(a.z, c.z, s);
        s = fmaf(a.w, c.w, s);
    }
    logits[i] = s;
    yhat[i] = 1.0f / (1.0f + __expf(-s));
}

// ---------------------------------------------------------------------------
// Kernel 7: mean BCE-with-logits reduction (single block)
// ---------------------------------------------------------------------------
__global__ __launch_bounds__(256) void k_loss(
    const float* __restrict__ logits, const float* __restrict__ target,
    float* __restrict__ out_loss)
{
    __shared__ float red[256];
    float sum = 0.0f;
    for (int i = threadIdx.x; i < B_ * Y_; i += 256) {
        const float z = logits[i];
        const float t = target[i];
        sum += fmaxf(z, 0.0f) - z * t + log1pf(__expf(-fabsf(z)));
    }
    red[threadIdx.x] = sum;
    __syncthreads();
    for (int o = 128; o > 0; o >>= 1) {
        if (threadIdx.x < o) red[threadIdx.x] += red[threadIdx.x + o];
        __syncthreads();
    }
    if (threadIdx.x == 0) *out_loss = red[0] / (float)(B_ * Y_);
}

// ---------------------------------------------------------------------------
extern "C" void kernel_launch(void* const* d_in, const int* in_sizes, int n_in,
                              void* d_out, int out_size, void* d_ws, size_t ws_size,
                              hipStream_t stream)
{
    const int*   x      = (const int*)  d_in[0];
    const float* target = (const float*)d_in[1];
    const float* emb    = (const float*)d_in[2];
    const float* convw  = (const float*)d_in[3];
    const float* convb  = (const float*)d_in[4];
    const float* U      = (const float*)d_in[5];
    const float* fw     = (const float*)d_in[6];
    const float* fb     = (const float*)d_in[7];

    // Output layout: yhat [B*Y] | loss [1] | alpha [B*Y*L]
    float* out   = (float*)d_out;
    float* yhat  = out;
    float* loss  = out + (size_t)B_ * Y_;
    float* alpha = loss + 1;

    // Workspace carve-out (≈45 MB): h16(padded) | U16(padded) | m | logits
    char* ws = (char*)d_ws;
    size_t off = 0;
    auto carve = [&](size_t bytes) -> void* {
        void* p = ws + off;
        off = (off + bytes + 255) & ~(size_t)255;
        return p;
    };
    _Float16* h16 = (_Float16*)carve((size_t)B_ * LP * D_ * sizeof(_Float16));
    _Float16* U16 = (_Float16*)carve((size_t)YP * D_ * sizeof(_Float16));
    float*    mws = (float*)   carve((size_t)B_ * Y_ * D_ * sizeof(float));
    float*    lg  = (float*)   carve((size_t)B_ * Y_ * sizeof(float));

    const int ytiles = (Y_ + 15) / 16;           // 558

    // 1) embedding + conv + tanh -> h16 (zero-padded rows up to LP)
    hipLaunchKernelGGL(k_embed_conv, dim3(LP / TL, B_), dim3(128),
                       0, stream, x, emb, convw, convb, h16);
    // 2) U -> f16 (zero-padded rows up to YP)
    hipLaunchKernelGGL(k_u_to_f16, dim3((YP * D_ + 255) / 256), dim3(256),
                       0, stream, U, U16);
    // 3) scores (raw) -> alpha slot of d_out   [WMMA, 2x4 tiles/wave]
    hipLaunchKernelGGL(k_scores, dim3(LP / 64, (YP / 32 + 3) / 4, B_),
                       dim3(128), 0, stream, U16, h16, alpha);
    // 4) softmax in place
    hipLaunchKernelGGL(k_softmax, dim3((B_ * Y_ + 3) / 4), dim3(128),
                       0, stream, alpha);
    // 5) m = alpha x h   [WMMA, double-buffered transposed LDS h]
    hipLaunchKernelGGL(k_mpool, dim3((ytiles + 3) / 4, B_),
                       dim3(128), 0, stream, alpha, h16, mws);
    // 6) logits + sigmoid
    hipLaunchKernelGGL(k_logits, dim3((B_ * Y_ + 255) / 256), dim3(256),
                       0, stream, mws, fw, fb, lg, yhat);
    // 7) loss
    hipLaunchKernelGGL(k_loss, dim3(1), dim3(256), 0, stream, lg, target, loss);
}
